// MinGRU_53730040873452
// MI455X (gfx1250) — matrix operand order repaired and checked by reading, compiled, not run on hardware
//
#include <hip/hip_runtime.h>

#ifndef __has_builtin
#define __has_builtin(x) 0
#endif

#if __has_builtin(__builtin_amdgcn_global_load_async_to_lds_b128)
#define USE_ASYNC_LDS 1
#else
#define USE_ASYNC_LDS 0
#endif

#define AS1 __attribute__((address_space(1)))
#define AS3 __attribute__((address_space(3)))

// ---------------------------------------------------------------------------
// Types for CDNA5 WMMA (wave32): v_wmma_f32_16x16x32_bf16
// ---------------------------------------------------------------------------
typedef __attribute__((ext_vector_type(16))) __bf16 v16bf;
typedef __attribute__((ext_vector_type(8)))  float  v8f;
typedef __attribute__((ext_vector_type(4)))  unsigned int u32x4;
// matches the async-to-LDS builtin's parameter type exactly:
typedef int v4i_gcc __attribute__((vector_size(16)));

union FragAB {
  v16bf v;
  u32x4 q[2];
};

__device__ __forceinline__ void wait_async_zero() {
#if __has_builtin(__builtin_amdgcn_s_wait_asynccnt)
  __builtin_amdgcn_s_wait_asynccnt(0);
#else
  asm volatile("s_wait_asynccnt 0x0" ::: "memory");
#endif
}

__device__ __forceinline__ float splusf(float x) {
  // stable softplus
  return fmaxf(x, 0.0f) + log1pf(expf(-fabsf(x)));
}
__device__ __forceinline__ float log_g_f(float x) {
  return (x >= 0.0f) ? logf(x + 0.5f) : -splusf(-x);
}
__device__ __forceinline__ float gelu_exact(float x) {
  return 0.5f * x * (1.0f + erff(x * 0.70710678118654752440f));
}

// ---------------------------------------------------------------------------
// f32 -> bf16 convert
// ---------------------------------------------------------------------------
__global__ void f32_to_bf16_kernel(const float* __restrict__ in,
                                   __bf16* __restrict__ out, size_t n) {
  size_t i = (size_t)blockIdx.x * blockDim.x + threadIdx.x;
  if (i < n) out[i] = (__bf16)in[i];
}

// ---------------------------------------------------------------------------
// Tiled WMMA GEMM:  C[M,N] = epilogue( A[M,K](bf16,row) * W[N,K](bf16,row)^T )
// Block tile 128x128x32, 256 threads = 8 waves, each wave -> 64x32 output.
// Tiles staged into LDS with GLOBAL_LOAD_ASYNC_TO_LDS_B128 when available.
// MODE 0: C32 = acc                     (non-temporal store: streaming output)
// MODE 1: C32 = gelu(acc + bias[n]); C16 = bf16(C32)
// MODE 2: C32 = acc + resid[m,n]
// M,N divisible by 128; K divisible by 32.
// ---------------------------------------------------------------------------
#define BM 128
#define BN 128
#define BKK 32
#define LDST 40  // LDS row stride in bf16 elements (padded: 80B rows)

template <int MODE>
__global__ __launch_bounds__(256) void gemm_bf16_wmma(
    const __bf16* __restrict__ A, const __bf16* __restrict__ W,
    float* __restrict__ C32, __bf16* __restrict__ C16,
    const float* __restrict__ bias, const float* __restrict__ resid,
    int M, int N, int K) {
  __shared__ __align__(16) __bf16 lA[2][BM * LDST];
  __shared__ __align__(16) __bf16 lB[2][BN * LDST];

  const int tid  = threadIdx.x;
  const int lane = tid & 31;
  const int wv   = tid >> 5;
  const long bm  = (long)blockIdx.y * BM;
  const long bn  = (long)blockIdx.x * BN;
  const int wM   = (wv >> 2) * 64;  // 0 or 64
  const int wN   = (wv & 3) * 32;   // 0..96
  const int r    = lane & 15;
  const int hi   = lane >> 4;

  // this thread's two 16B staging chunks within a 128x32 tile
  const int c0row = tid >> 2;             // chunk 0: rows 0..63
  const int c0col = (tid & 3) << 3;
  const int c1row = (tid + 256) >> 2;     // chunk 1: rows 64..127
  const int c1col = c0col;

  v8f acc[4][2];
#pragma unroll
  for (int m = 0; m < 4; ++m)
#pragma unroll
    for (int n = 0; n < 2; ++n)
#pragma unroll
      for (int i = 0; i < 8; ++i) acc[m][n][i] = 0.0f;

  // ---- stage k-tile k0 into LDS buffer `dst` ----
  auto stage_tile = [&](int k0, int dst) {
#if USE_ASYNC_LDS
    __builtin_amdgcn_global_load_async_to_lds_b128(
        (AS1 v4i_gcc*)&A[(bm + c0row) * (long)K + k0 + c0col],
        (AS3 v4i_gcc*)&lA[dst][c0row * LDST + c0col], 0, 0);
    __builtin_amdgcn_global_load_async_to_lds_b128(
        (AS1 v4i_gcc*)&A[(bm + c1row) * (long)K + k0 + c1col],
        (AS3 v4i_gcc*)&lA[dst][c1row * LDST + c1col], 0, 0);
    __builtin_amdgcn_global_load_async_to_lds_b128(
        (AS1 v4i_gcc*)&W[(bn + c0row) * (long)K + k0 + c0col],
        (AS3 v4i_gcc*)&lB[dst][c0row * LDST + c0col], 0, 0);
    __builtin_amdgcn_global_load_async_to_lds_b128(
        (AS1 v4i_gcc*)&W[(bn + c1row) * (long)K + k0 + c1col],
        (AS3 v4i_gcc*)&lB[dst][c1row * LDST + c1col], 0, 0);
#else
    *(u32x4*)&lA[dst][c0row * LDST + c0col] =
        *(const u32x4*)&A[(bm + c0row) * (long)K + k0 + c0col];
    *(u32x4*)&lA[dst][c1row * LDST + c1col] =
        *(const u32x4*)&A[(bm + c1row) * (long)K + k0 + c1col];
    *(u32x4*)&lB[dst][c0row * LDST + c0col] =
        *(const u32x4*)&W[(bn + c0row) * (long)K + k0 + c0col];
    *(u32x4*)&lB[dst][c1row * LDST + c1col] =
        *(const u32x4*)&W[(bn + c1row) * (long)K + k0 + c1col];
#endif
  };

  stage_tile(0, 0);
#if USE_ASYNC_LDS
  wait_async_zero();
#endif
  __syncthreads();

  const int nk = K / BKK;
  for (int kt = 0; kt < nk; ++kt) {
    const int buf = kt & 1;
    // prefetch two tiles ahead into L2 (global_prefetch_b8)
    if (kt + 2 < nk) {
      const int kp = (kt + 2) * BKK;
      __builtin_prefetch(&A[(bm + c0row) * (long)K + kp + c0col], 0, 3);
      __builtin_prefetch(&A[(bm + c1row) * (long)K + kp + c1col], 0, 3);
      __builtin_prefetch(&W[(bn + c0row) * (long)K + kp + c0col], 0, 3);
      __builtin_prefetch(&W[(bn + c1row) * (long)K + kp + c1col], 0, 3);
    }
    if (kt + 1 < nk) stage_tile((kt + 1) * BKK, buf ^ 1);

    // fragments (ISA 7.12.2 bf16 layouts)
    FragAB fa[4], fb[2];
    const int kbA = hi * 8;   // lanes 0-15: K 0-7 & 16-23 ; lanes 16-31: 8-15 & 24-31
    const int kbB = hi * 8;   // lanes 0-15: K 0-7 & 16-23 ; lanes 16-31: 8-15 & 24-31
#pragma unroll
    for (int m = 0; m < 4; ++m) {
      const __bf16* ap = &lA[buf][(wM + m * 16 + r) * LDST];
      fa[m].q[0] = *(const u32x4*)(ap + kbA);
      fa[m].q[1] = *(const u32x4*)(ap + kbA + 16);
    }
#pragma unroll
    for (int n = 0; n < 2; ++n) {
      const __bf16* bp = &lB[buf][(wN + n * 16 + r) * LDST];
      fb[n].q[0] = *(const u32x4*)(bp + kbB);
      fb[n].q[1] = *(const u32x4*)(bp + kbB + 16);
    }
#pragma unroll
    for (int m = 0; m < 4; ++m)
#pragma unroll
      for (int n = 0; n < 2; ++n)
        acc[m][n] = __builtin_amdgcn_wmma_f32_16x16x32_bf16(
            false, fa[m].v, false, fb[n].v, (short)0, acc[m][n], false, false);

#if USE_ASYNC_LDS
    if (kt + 1 < nk) wait_async_zero();
#endif
    __syncthreads();
  }

  // epilogue: C/D layout: VGPR i -> row = i + 8*hi, col = lane&15
#pragma unroll
  for (int m = 0; m < 4; ++m) {
#pragma unroll
    for (int n = 0; n < 2; ++n) {
      const long col = bn + wN + n * 16 + r;
#pragma unroll
      for (int i = 0; i < 8; ++i) {
        const long row = bm + wM + m * 16 + hi * 8 + i;
        const long off = row * (long)N + col;
        float x = acc[m][n][i];
        if (MODE == 1) {
          x += bias[col];
          x = gelu_exact(x);
          C32[off] = x;
          C16[off] = (__bf16)x;
        } else if (MODE == 2) {
          C32[off] = x + resid[off];
        } else {
          // streaming output (hg: 256MB, written once, read once)
          __builtin_nontemporal_store(x, &C32[off]);
        }
      }
    }
  }
}

// ---------------------------------------------------------------------------
// MinGRU log-space scan along T. One thread per (batch, channel).
// hg: [B, T, 2*DI] f32 (hidden = [:DI], gate = [DI:]) ; v: [B, T, DI] bf16
// ---------------------------------------------------------------------------
__global__ void scan_kernel(const float* __restrict__ hg, __bf16* __restrict__ v,
                            int T, int DI) {
  const int idx = blockIdx.x * blockDim.x + threadIdx.x;  // 0..B*DI-1
  const int b = idx / DI;
  const int d = idx - b * DI;
  const float* base = hg + (size_t)b * T * 2 * DI;
  __bf16* vp = v + (size_t)b * T * DI + d;
  float lh = -1e30f;
  for (int t = 0; t < T; ++t) {
    const float* p = base + (size_t)t * 2 * DI;
    const float hid  = __builtin_nontemporal_load(&p[d]);
    const float gate = __builtin_nontemporal_load(&p[DI + d]);
    const float lc = -splusf(gate);                    // log a_t
    const float lv = -splusf(-gate) + log_g_f(hid);    // log(z_t * g(h~))
    const float a = lc + lh;
    const float mx = fmaxf(a, lv);
    lh = mx + log1pf(expf(-fabsf(a - lv)));            // logaddexp
    vp[(size_t)t * DI] = (__bf16)expf(lh);
  }
}

// ---------------------------------------------------------------------------
// RMS norm over H=512, one wave (32 lanes x 16 values) per row.
// out32/out16 get the normalized row (f32 + bf16 copies).
// ---------------------------------------------------------------------------
__global__ void rmsnorm_kernel(const float* __restrict__ in,
                               const float* __restrict__ rw,
                               float* __restrict__ out32,
                               __bf16* __restrict__ out16, int M, int H) {
  const int wave = (int)((blockIdx.x * (long)blockDim.x + threadIdx.x) >> 5);
  const int lane = threadIdx.x & 31;
  if (wave >= M) return;
  const float* rp = in + (size_t)wave * H;
  float vals[16];
  float ss = 0.0f;
#pragma unroll
  for (int j = 0; j < 16; ++j) {
    vals[j] = rp[lane + j * 32];
    ss += vals[j] * vals[j];
  }
#pragma unroll
  for (int off = 16; off >= 1; off >>= 1) ss += __shfl_xor(ss, off, 32);
  const float scale = rsqrtf(ss / (float)H + 1.1920929e-07f);
#pragma unroll
  for (int j = 0; j < 16; ++j) {
    const int c = lane + j * 32;
    const float o = vals[j] * scale * rw[c];
    out32[(size_t)wave * H + c] = o;
    out16[(size_t)wave * H + c] = (__bf16)o;
  }
}

// ---------------------------------------------------------------------------
// Decoder: logits[M,16] via WMMA (one wave per 16 rows), value head via dot.
// ---------------------------------------------------------------------------
__global__ void decoder_kernel(const __bf16* __restrict__ hb,
                               const float* __restrict__ h32,
                               const __bf16* __restrict__ dwb,
                               const float* __restrict__ dec_b,
                               const float* __restrict__ val_w,
                               const float* __restrict__ val_b,
                               float* __restrict__ logits,
                               float* __restrict__ values, int M, int K) {
  const int wave = (int)((blockIdx.x * (long)blockDim.x + threadIdx.x) >> 5);
  const int lane = threadIdx.x & 31;
  const long row0 = (long)wave * 16;
  if (row0 >= M) return;
  const int r = lane & 15;
  const int hi = lane >> 4;

  v8f acc;
#pragma unroll
  for (int i = 0; i < 8; ++i) acc[i] = 0.0f;

  const int kbA = hi * 8;
  const int kbB = hi * 8;
  for (int k0 = 0; k0 < K; k0 += 32) {
    FragAB fa, fb;
    const __bf16* ap = &hb[(row0 + r) * (long)K + k0];
    fa.q[0] = *(const u32x4*)(ap + kbA);
    fa.q[1] = *(const u32x4*)(ap + kbA + 16);
    const __bf16* bp = &dwb[(long)r * K + k0];
    fb.q[0] = *(const u32x4*)(bp + kbB);
    fb.q[1] = *(const u32x4*)(bp + kbB + 16);
    acc = __builtin_amdgcn_wmma_f32_16x16x32_bf16(false, fa.v, false, fb.v,
                                                  (short)0, acc, false, false);
  }
#pragma unroll
  for (int i = 0; i < 8; ++i) {
    const long rr = row0 + hi * 8 + i;
    logits[rr * 16 + r] = acc[i] + dec_b[r];
  }
  // value head: lane (r, hi) covers half a row each
  const float* hrow = &h32[(row0 + r) * (long)K];
  float s = 0.0f;
  const int kk0 = hi * (K / 2);
  for (int k = 0; k < K / 2; ++k) s += hrow[kk0 + k] * val_w[kk0 + k];
  s += __shfl_xor(s, 16, 32);
  if (lane < 16) values[row0 + r] = s + val_b[0];
}

// ---------------------------------------------------------------------------
// Host orchestration
// ---------------------------------------------------------------------------
extern "C" void kernel_launch(void* const* d_in, const int* in_sizes, int n_in,
                              void* d_out, int out_size, void* d_ws,
                              size_t ws_size, hipStream_t stream) {
  (void)in_sizes; (void)n_in; (void)out_size; (void)ws_size;
  const int Bq = 32, T = 1024, OBS = 512, H = 512, DI = 1024, L = 2, NA = 16;
  const long M = (long)Bq * T;  // 32768

  const float* x     = (const float*)d_in[0];
  const float* enc_w = (const float*)d_in[1];
  const float* enc_b = (const float*)d_in[2];
  const float* w_hg  = (const float*)d_in[3];
  const float* w_out = (const float*)d_in[4];
  const float* rms_w = (const float*)d_in[5];
  const float* dec_w = (const float*)d_in[6];
  const float* dec_b = (const float*)d_in[7];
  const float* val_w = (const float*)d_in[8];
  const float* val_b = (const float*)d_in[9];

  char* ws = (char*)d_ws;
  size_t off = 0;
  auto alloc = [&](size_t bytes) -> void* {
    void* p = ws + off;
    off += (bytes + 255) & ~(size_t)255;
    return p;
  };

  __bf16* xb    = (__bf16*)alloc((size_t)M * OBS * 2);
  __bf16* ewb   = (__bf16*)alloc((size_t)H * OBS * 2);
  __bf16* hgwb  = (__bf16*)alloc((size_t)L * 2 * DI * H * 2);
  __bf16* owb   = (__bf16*)alloc((size_t)L * H * DI * 2);
  __bf16* dwb   = (__bf16*)alloc((size_t)NA * H * 2);
  float*  h32   = (float*)alloc((size_t)M * H * 4);
  __bf16* hb    = (__bf16*)alloc((size_t)M * H * 2);
  float*  hg32  = (float*)alloc((size_t)M * 2 * DI * 4);
  __bf16* vb    = (__bf16*)alloc((size_t)M * DI * 2);
  float*  pre32 = hg32;  // hg dead after scan -> reuse for pre-norm activations

  auto cvt = [&](const float* src, __bf16* dst, size_t n) {
    f32_to_bf16_kernel<<<(unsigned)((n + 255) / 256), 256, 0, stream>>>(src, dst, n);
  };
  cvt(x, xb, (size_t)M * OBS);
  cvt(enc_w, ewb, (size_t)H * OBS);
  cvt(w_hg, hgwb, (size_t)L * 2 * DI * H);
  cvt(w_out, owb, (size_t)L * H * DI);
  cvt(dec_w, dwb, (size_t)NA * H);

  // encoder: h = gelu(x @ enc_w^T + enc_b)
  gemm_bf16_wmma<1><<<dim3(H / BN, M / BM), 256, 0, stream>>>(
      xb, ewb, h32, hb, enc_b, nullptr, (int)M, H, OBS);

  for (int l = 0; l < L; ++l) {
    // hg = h @ w_hg^T  (N = 2*DI = 2048)
    gemm_bf16_wmma<0><<<dim3(2 * DI / BN, M / BM), 256, 0, stream>>>(
        hb, hgwb + (size_t)l * 2 * DI * H, hg32, nullptr, nullptr, nullptr,
        (int)M, 2 * DI, H);
    // sequential log-space scan -> v = exp(log_h) (bf16)
    scan_kernel<<<(Bq * DI) / 256, 256, 0, stream>>>(hg32, vb, T, DI);
    // out_pre = v @ w_out^T + h   (residual fused)
    gemm_bf16_wmma<2><<<dim3(H / BN, M / BM), 256, 0, stream>>>(
        vb, owb + (size_t)l * H * DI, pre32, nullptr, nullptr, h32, (int)M, H,
        DI);
    // h = rmsnorm(out_pre) * rms_w
    rmsnorm_kernel<<<(unsigned)(M * 32 / 256), 256, 0, stream>>>(
        pre32, rms_w + (size_t)l * H, h32, hb, (int)M, H);
  }

  float* logits = (float*)d_out;
  float* values = logits + M * NA;
  decoder_kernel<<<(unsigned)(M / 16 / 8), 256, 0, stream>>>(
      hb, h32, dwb, dec_b, val_w, val_b, logits, values, (int)M, H);
}
